// SigNet_2087354106687
// MI455X (gfx1250) — compile-verified, matrix operand order checked by reading, not run on hardware
//
#include <hip/hip_runtime.h>
#include <hip/hip_bf16.h>

#define B_N   128
#define S_N   1024
#define C_IN  7
#define C_N   8
#define L2_N  64
#define L3_N  512
#define SIGC  4680
#define DOUT  256

typedef __attribute__((ext_vector_type(2)))  float v2f;
typedef __attribute__((ext_vector_type(8)))  float v8f;

// ---------------------------------------------------------------------------
// Kernel 1: depth-4 signature scan. One workgroup per batch, 256 threads.
// Levels 1-3 (584 floats) live in LDS; level 4 (4096 floats) lives entirely in
// registers (16 per thread). For thread t, the level-4 chunk [16t,16t+16) and
// the level-3 chunk [2t,2t+2) share the same (i,j,k) prefix, so per step each
// thread needs only v[0..8), s1[i], s2[ij] and its own two s3 entries.
// ---------------------------------------------------------------------------
__global__ __launch_bounds__(256) void sig_scan_kernel(const float* __restrict__ inp,
                                                       float* __restrict__ sig)
{
    __shared__ float s1[C_N];
    __shared__ float s2[L2_N];
    __shared__ float s3[L3_N];
    __shared__ float vsh[C_N];

    const int b = blockIdx.x;
    const int t = threadIdx.x;

    float s4r[16];
#pragma unroll
    for (int q = 0; q < 16; ++q) s4r[q] = 0.0f;

    if (t < C_N)  s1[t] = 0.0f;
    if (t < L2_N) s2[t] = 0.0f;
    s3[2 * t]     = 0.0f;
    s3[2 * t + 1] = 0.0f;

    // shared (i,j,k) prefix of this thread's level-3 and level-4 chunks
    const int i4  = t >> 5;          // idx4>>9  == (2t)>>6
    const int j4  = (t >> 2) & 7;    // (idx4>>6)&7
    const int k4  = (t << 1) & 7;    // even; chunk covers k4 and k4+1
    const int ij4 = i4 * 8 + j4;

    const float dt = 1.0f / 1023.0f;
    const float* inb = inp + (size_t)b * S_N * C_IN;
    float prev = 0.0f;               // previous sample for this thread's channel

    for (int s = 0; s < S_N; ++s) {
        // ---- phase A: threads 0..7 produce the increment v ----------------
        if (t == 0) {
            vsh[0] = (s == 0) ? 0.0f : dt;           // time channel
        } else if (t < C_N) {
            float cur = inb[s * C_IN + (t - 1)];
            vsh[t] = cur - prev;                     // basepoint=0 => prev=0 at s=0
            prev = cur;
        }
        if (t < C_IN && (s + 8) < S_N)
            __builtin_prefetch(&inb[(s + 8) * C_IN + t], 0, 1);  // global_prefetch_b8
        __syncthreads();

        // ---- phase B: read all old state needed into registers ------------
        float v[8];
#pragma unroll
        for (int c = 0; c < 8; ++c) v[c] = vsh[c];

        const float a1  = s1[i4];
        const float a2  = s2[ij4];
        const float o30 = s3[2 * t];
        const float o31 = s3[2 * t + 1];
        float o2 = 0.0f, a1b = 0.0f;
        if (t < L2_N) { o2 = s2[t]; a1b = s1[t >> 3]; }
        const float o1 = (t < C_N) ? s1[t] : 0.0f;
        __syncthreads();

        // ---- phase C: compute + write -------------------------------------
        const float vi  = v[i4];
        const float vj  = v[j4];
        const float vij = vi * vj;

        // level 4: new4[i,j,k,l] = old + (vij*vk/24 + s1_i*vj*vk/6 + s2_ij*vk/2 + s3_ijk) * v_l
#pragma unroll
        for (int kk = 0; kk < 2; ++kk) {
            const float vk = v[k4 + kk];
            const float o3 = (kk == 0) ? o30 : o31;
            const float coef = vk * (vij * (1.0f / 24.0f)
                                     + a1 * vj * (1.0f / 6.0f)
                                     + a2 * 0.5f) + o3;
#pragma unroll
            for (int l = 0; l < 8; ++l)
                s4r[kk * 8 + l] += coef * v[l];
        }

        // level 3: new3[i,j,k] = old + (vij/6 + s1_i*vj/2 + s2_ij) * v_k
        const float c3 = vij * (1.0f / 6.0f) + a1 * vj * 0.5f + a2;
        s3[2 * t]     = o30 + c3 * v[k4];
        s3[2 * t + 1] = o31 + c3 * v[k4 + 1];

        // level 2: new2[i,j] = old + (v_i/2 + s1_i) * v_j
        if (t < L2_N) {
            const int i2 = t >> 3, j2 = t & 7;
            s2[t] = o2 + (v[i2] * 0.5f + a1b) * v[j2];
        }
        // level 1
        if (t < C_N) s1[t] = o1 + v[t];
    }
    __syncthreads();

    // concat [s1 | s2 | s3 | s4] -> sig[b, 0:4680]
    float* sb = sig + (size_t)b * SIGC;
    if (t < C_N)  sb[t] = s1[t];
    if (t < L2_N) sb[8 + t] = s2[t];
    sb[72 + 2 * t]     = s3[2 * t];
    sb[72 + 2 * t + 1] = s3[2 * t + 1];
#pragma unroll
    for (int q = 0; q < 16; ++q) sb[584 + t * 16 + q] = s4r[q];
}

// ---------------------------------------------------------------------------
// Kernel 2: out = sig(128x4680) @ W^T(4680x256) + b, via fp32 WMMA.
// One wave per 16x16 output tile; 8 waves per block, 16 blocks = 128 tiles.
// A 16x4 layout: lanes 0-15 hold M, vgpr pair = K{0,1}; lanes 16-31 = K{2,3}.
// B 4x16 layout mirrors it with N across lanes.
// ---------------------------------------------------------------------------
#if __has_builtin(__builtin_amdgcn_wmma_f32_16x16x4_f32)
__global__ __launch_bounds__(256) void sig_gemm(const float* __restrict__ sig,
                                                const float* __restrict__ W,
                                                const float* __restrict__ bias,
                                                float* __restrict__ out)
{
    const int lane = threadIdx.x & 31;
    const int wave = threadIdx.x >> 5;
    const int tile = blockIdx.x * 8 + wave;   // 0..127
    const int tm = tile >> 4;                 // 0..7   (M tile of 16)
    const int tn = tile & 15;                 // 0..15  (N tile of 16)

    const int mr    = tm * 16 + (lane & 15);
    const int nr    = tn * 16 + (lane & 15);
    const int khalf = (lane >> 4) << 1;       // 0 or 2

    const float* ap = sig + (size_t)mr * SIGC + khalf;
    const float* bp = W   + (size_t)nr * SIGC + khalf;   // B[k][n] = W[n][k]

    v8f acc = {};
    for (int k0 = 0; k0 < SIGC; k0 += 4) {    // 4680 % 4 == 0
        v2f av = *(const v2f*)(ap + k0);
        v2f bv = *(const v2f*)(bp + k0);
        acc = __builtin_amdgcn_wmma_f32_16x16x4_f32(false, av, false, bv,
                                                    (short)0, acc, false, false);
    }

    const int mbase = tm * 16 + ((lane >> 4) << 3);
    const int ncol  = tn * 16 + (lane & 15);
    const float bb = bias[ncol];
#pragma unroll
    for (int r = 0; r < 8; ++r)
        out[(size_t)(mbase + r) * DOUT + ncol] = acc[r] + bb;
}
#else
// Fallback: codegen-confirmed bf16 WMMA (f32 accumulate), K-step 32, guarded tail.
typedef __attribute__((ext_vector_type(16))) __bf16 v16bf;
__global__ __launch_bounds__(256) void sig_gemm(const float* __restrict__ sig,
                                                const float* __restrict__ W,
                                                const float* __restrict__ bias,
                                                float* __restrict__ out)
{
    const int lane = threadIdx.x & 31;
    const int wave = threadIdx.x >> 5;
    const int tile = blockIdx.x * 8 + wave;
    const int tm = tile >> 4;
    const int tn = tile & 15;

    const int mr  = tm * 16 + (lane & 15);
    const int nr  = tn * 16 + (lane & 15);
    const int off = (lane >> 4) << 3;         // 0 or 8

    const float* ap = sig + (size_t)mr * SIGC;
    const float* bp = W   + (size_t)nr * SIGC;

    v8f acc = {};
    for (int k0 = 0; k0 < SIGC; k0 += 32) {
        v16bf av, bv;
#pragma unroll
        for (int e = 0; e < 8; ++e) {
            int ka = k0 + off + e;
            int kb = k0 + off + 16 + e;
            av[e]     = (__bf16)((ka < SIGC) ? ap[ka] : 0.0f);
            av[e + 8] = (__bf16)((kb < SIGC) ? ap[kb] : 0.0f);
            bv[e]     = (__bf16)((ka < SIGC) ? bp[ka] : 0.0f);
            bv[e + 8] = (__bf16)((kb < SIGC) ? bp[kb] : 0.0f);
        }
        acc = __builtin_amdgcn_wmma_f32_16x16x32_bf16(false, av, false, bv,
                                                      (short)0, acc, false, false);
    }
    const int mbase = tm * 16 + ((lane >> 4) << 3);
    const int ncol  = tn * 16 + (lane & 15);
    const float bb = bias[ncol];
#pragma unroll
    for (int r = 0; r < 8; ++r)
        out[(size_t)(mbase + r) * DOUT + ncol] = acc[r] + bb;
}
#endif

extern "C" void kernel_launch(void* const* d_in, const int* in_sizes, int n_in,
                              void* d_out, int out_size, void* d_ws, size_t ws_size,
                              hipStream_t stream)
{
    (void)in_sizes; (void)n_in; (void)out_size; (void)ws_size;
    const float* inp  = (const float*)d_in[0];   // (128,1024,7) f32
    const float* W    = (const float*)d_in[1];   // (256,4680)   f32
    const float* bias = (const float*)d_in[2];   // (256,)       f32
    float* out = (float*)d_out;                  // (128,256)    f32
    float* sig = (float*)d_ws;                   // (128,4680)   f32 scratch, 2.4 MB

    sig_scan_kernel<<<B_N, 256, 0, stream>>>(inp, sig);
    sig_gemm<<<16, 256, 0, stream>>>(sig, W, bias, out);
}
